// FastPitch_13881334301178
// MI455X (gfx1250) — matrix operand order repaired and checked by reading, not compiled, not run on hardware
//
#include <hip/hip_runtime.h>
#include <stdint.h>

typedef unsigned short u16;
typedef __attribute__((ext_vector_type(8)))  float  v8f;
typedef __attribute__((ext_vector_type(16))) __bf16 v16bf;
typedef __attribute__((ext_vector_type(8)))  u16    v8u;
typedef __attribute__((ext_vector_type(16))) u16    v16u;

#define B_    16
#define T_    256
#define D_    384
#define M_    2048
#define FILT_ 256
#define NMEL_ 80
#define PACE_ 1.0f

__device__ __forceinline__ u16 f2b(float f) {
  unsigned u = __float_as_uint(f);
  u += 0x7fffu + ((u >> 16) & 1u);          // round-to-nearest-even to bf16
  return (u16)(u >> 16);
}
__device__ __forceinline__ float b2f(u16 h) {
  return __uint_as_float(((unsigned)h) << 16);
}

// ---------- generic f32 -> bf16 bit convert ----------
__global__ void k_cvt(const float* __restrict__ s, u16* __restrict__ d, int n) {
  int i = blockIdx.x * blockDim.x + threadIdx.x;
  if (i < n) d[i] = f2b(s[i]);
}

// ---------- conv weight convert + permute: [O,Cin,3] f32 -> bf16 with
// k_new = cin_block*96 + shift*32 + c  (cin = cin_block*32 + c) ----------
__global__ void k_cvt_w(const float* __restrict__ s, u16* __restrict__ d, int Cin, int n) {
  int i = blockIdx.x * blockDim.x + threadIdx.x;
  if (i >= n) return;
  int row = Cin * 3;
  int o   = i / row;
  int pos = i - o * row;
  int cb  = pos / 96;
  int rem = pos - cb * 96;
  int sft = rem >> 5;
  int c   = rem & 31;
  int cin = (cb << 5) + c;
  d[i] = f2b(s[(o * Cin + cin) * 3 + sft]);
}

// ---------- enc_out [B,T,D] * mask -> X0 [B,D,T] bf16 ----------
__global__ void k_transpose_mask(const float* __restrict__ enc, const float* __restrict__ mask,
                                 u16* __restrict__ x0) {
  int i = blockIdx.x * blockDim.x + threadIdx.x;  // over B*D*T
  int t = i % T_;
  int c = (i / T_) % D_;
  int b = i / (T_ * D_);
  float v = enc[(b * T_ + t) * D_ + c] * mask[b * T_ + t];
  x0[i] = f2b(v);
}

// ---------- fused Conv1d(K=3)+bias+ReLU+LayerNorm(channels=256), WMMA bf16 ----------
// X: [B,Cin,T] bf16 bits; W: [256, Cin*3] bf16 bits in permuted k ordering;
// bias/gamma/beta: f32[256]; out: [B,256,T] bf16 bits.
__global__ __launch_bounds__(512) void k_conv_ln(const u16* __restrict__ X, const u16* __restrict__ W,
                                                 const float* __restrict__ bias,
                                                 const float* __restrict__ gamma,
                                                 const float* __restrict__ beta,
                                                 u16* __restrict__ out, int Cin) {
  const int Kdim = Cin * 3;
  const int nCB  = Cin >> 5;                  // 32-channel blocks
  const int tid  = threadIdx.x;
  const int wave = tid >> 5, lane = tid & 31;
  const int half = lane >> 4, n = lane & 15;  // n doubles as A-row m and B-col n
  const int b  = blockIdx.x >> 4;
  const int t0 = (blockIdx.x & 15) << 4;

  __shared__ alignas(64) u16 winT[18 * 32];   // [j = t-offset 0..17][c = channel 0..31]
  __shared__ float rsum[16], rsq[16];
  if (tid < 16) { rsum[tid] = 0.f; rsq[tid] = 0.f; }

  const u16* Wrow = W + (size_t)(16 * wave + n) * Kdim;
  const u16* Xb   = X + (size_t)b * Cin * T_;

  // staging coords: interior (j=1..16) is one unguarded load per thread
  const int sj = 1 + (tid >> 5);   // 1..16
  const int sc = tid & 31;         // channel-in-block

  v8f acc = {};
  for (int cb = 0; cb < nCB; ++cb) {
    const int cin0 = cb << 5;
    const u16* Xblk = Xb + (size_t)(cin0) * T_;
    // interior columns: winT[j][c] = X[cin0+c, t0+j-1], t in [t0, t0+15], always valid
    winT[sj * 32 + sc] = Xblk[sc * T_ + (t0 + sj - 1)];
    // boundary columns j=0 (t0-1) and j=17 (t0+16): only guards that exist
    if (tid < 32) {
      int t = t0 - 1;
      winT[tid] = (t >= 0) ? Xblk[tid * T_ + t] : (u16)0;
    } else if (tid < 64) {
      int c = tid - 32, t = t0 + 16;
      winT[17 * 32 + c] = (t < T_) ? Xblk[c * T_ + t] : (u16)0;
    }
    const int kb = cb * 96;
    __builtin_prefetch(Wrow + kb + 96, 0, 0);                 // next weight block
    __builtin_prefetch(Xblk + 32 * T_ + t0, 0, 0);            // next input block
    __syncthreads();
#pragma unroll
    for (int s = 0; s < 3; ++s) {
      const int kk = kb + s * 32;
      v16u au = *(const v16u*)(Wrow + kk + 16 * half);
      // B[kl][n] = winT[n+s][kl]; lane reads 16 contiguous u16 (32B aligned)
      v16u bu = *(const v16u*)(winT + (n + s) * 32 + 16 * half);
      acc = __builtin_amdgcn_wmma_f32_16x16x32_bf16(false, __builtin_bit_cast(v16bf, au),
                                                    false, __builtin_bit_cast(v16bf, bu),
                                                    (short)0, acc, false, false);
    }
    __syncthreads();
  }

  // epilogue: bias + relu, then LayerNorm over 256 channels per position
  float s = 0.f, q = 0.f, vals[8];
#pragma unroll
  for (int r = 0; r < 8; ++r) {
    int c = 16 * wave + 8 * half + r;
    float v = acc[r] + bias[c];
    v = v > 0.f ? v : 0.f;
    vals[r] = v; s += v; q += v * v;
  }
  atomicAdd(&rsum[n], s);   // ds_add_f32
  atomicAdd(&rsq[n], q);
  __syncthreads();
  float mean = rsum[n] * (1.f / 256.f);
  float var  = rsq[n] * (1.f / 256.f) - mean * mean;
  float inv  = rsqrtf(var + 1e-5f);
  int p = t0 + n;
#pragma unroll
  for (int r = 0; r < 8; ++r) {
    int c = 16 * wave + 8 * half + r;
    float y = (vals[r] - mean) * inv * gamma[c] + beta[c];
    out[((size_t)b * 256 + c) * T_ + p] = f2b(y);
  }
}

// ---------- FC (256 -> 1) + mask; dur branch also emits clip(exp(x)-1, 0, 75) ----------
__global__ void k_fc(const u16* __restrict__ H, const float* __restrict__ fcw,
                     const float* __restrict__ fcb, const float* __restrict__ mask,
                     float* __restrict__ out_main, float* __restrict__ out_dur, int isDur) {
  int gw   = (blockIdx.x * blockDim.x + threadIdx.x) >> 5;  // one wave per (b,t)
  int lane = threadIdx.x & 31;
  if (gw >= B_ * T_) return;
  int b = gw / T_, t = gw % T_;
  float v = 0.f;
#pragma unroll
  for (int i = 0; i < 8; ++i) {
    int c = lane + 32 * i;
    v += b2f(H[((size_t)b * 256 + c) * T_ + t]) * fcw[c];
  }
#pragma unroll
  for (int off = 16; off > 0; off >>= 1) v += __shfl_xor(v, off);
  if (lane == 0) {
    v = (v + fcb[0]) * mask[b * T_ + t];
    out_main[gw] = v;
    if (isDur) {
      float d = expf(v) - 1.f;
      d = d < 0.f ? 0.f : (d > 75.f ? 75.f : d);
      out_dur[gw] = d;
    }
  }
}

// ---------- enc_bf16[b,t,d] = bf16(enc_out + conv1d(pitch_tgt; 1->D, K=3)) ----------
__global__ void k_encb(const float* __restrict__ enc, const float* __restrict__ pt,
                       const float* __restrict__ pw, const float* __restrict__ pb,
                       u16* __restrict__ encb) {
  int i = blockIdx.x * blockDim.x + threadIdx.x;  // B*T*D
  int d = i % D_;
  int t = (i / D_) % T_;
  int b = i / (D_ * T_);
  const float* row = pt + b * T_;
  float acc = pb[d];
  float w0 = pw[d * 3 + 0], w1 = pw[d * 3 + 1], w2 = pw[d * 3 + 2];
  if (t - 1 >= 0) acc += w0 * row[t - 1];
  acc += w1 * row[t];
  if (t + 1 < T_) acc += w2 * row[t + 1];
  encb[i] = f2b(enc[i] + acc);
}

// ---------- per-batch inclusive cumsum of reps; dec_len = min(sum, M) ----------
__global__ void k_scan(const int* __restrict__ dur, int* __restrict__ cums, int* __restrict__ declen) {
  __shared__ int s[256];
  int b = blockIdx.x, t = threadIdx.x;
  s[t] = (int)rintf((float)dur[b * T_ + t] / PACE_);
  __syncthreads();
  for (int off = 1; off < 256; off <<= 1) {
    int v = (t >= off) ? s[t - off] : 0;
    __syncthreads();
    s[t] += v;
    __syncthreads();
  }
  cums[b * T_ + t] = s[t];
  if (t == 255) declen[b] = s[255] < M_ ? s[255] : M_;
}

// ---------- searchsorted(right) + dec_mask ----------
__global__ void k_idx(const int* __restrict__ cums, const int* __restrict__ declen,
                      int* __restrict__ idxm, float* __restrict__ maskout) {
  int g = blockIdx.x * blockDim.x + threadIdx.x;  // B*M
  int b = g >> 11;
  int j = g & (M_ - 1);
  const int* c = cums + b * T_;
  int lo = 0, hi = T_;
  while (lo < hi) {
    int mid = (lo + hi) >> 1;
    if (c[mid] <= j) lo = mid + 1; else hi = mid;
  }
  int idx   = lo < T_ - 1 ? lo : T_ - 1;
  int valid = j < declen[b];
  idxm[g]    = valid ? idx : -1;   // -1 => masked row (gather zeros)
  maskout[g] = valid ? 1.f : 0.f;
}

// ---------- mel projection: gather rows of enc_bf16, GEMM vs proj_w (80x384), +bias ----------
__global__ __launch_bounds__(160) void k_mel(const u16* __restrict__ encb, const u16* __restrict__ Wp,
                                             const float* __restrict__ pbias,
                                             const int* __restrict__ idxm, float* __restrict__ mel) {
  const int tid  = threadIdx.x;
  const int wave = tid >> 5, lane = tid & 31;
  const int half = lane >> 4, n = lane & 15;
  int pos0 = blockIdx.x << 4;          // global (b*M + j) of first frame in tile
  int b    = pos0 >> 11;
  int j0   = pos0 & (M_ - 1);

  __shared__ alignas(64) u16 tileB[16 * 96];  // 16 frames x 96 K values
  __shared__ int sidx[16];
  if (tid < 16) sidx[tid] = idxm[b * M_ + j0 + tid];
  __syncthreads();

  // staging coords: 128 threads, 8 per frame row, 3 aligned 8B chunks each
  const int ns = tid >> 3;        // frame row (tid < 128)
  const int c8 = (tid & 7) << 2;  // first chunk element offset

  const u16* Wrow = Wp + (size_t)(16 * wave + n) * D_;
  v8f acc = {};
  for (int kb = 0; kb < D_; kb += 96) {
    if (tid < 128) {
      int row = sidx[ns];
      const u16* src = encb + (size_t)(b * T_ + (row >= 0 ? row : 0)) * D_ + kb + c8;
      u16* dst = tileB + ns * 96 + c8;
#pragma unroll
      for (int c3 = 0; c3 < 3; ++c3) {
        uint2 val; val.x = 0u; val.y = 0u;
        if (row >= 0) val = *(const uint2*)(src + c3 * 32);
        *(uint2*)(dst + c3 * 32) = val;
      }
    }
    __builtin_prefetch(Wrow + kb + 96, 0, 0);
    __syncthreads();
#pragma unroll
    for (int s = 0; s < 3; ++s) {
      const int kk = kb + s * 32;
      v16u au = *(const v16u*)(Wrow + kk + 16 * half);
      v16u bu = *(const v16u*)(tileB + n * 96 + s * 32 + 16 * half);
      acc = __builtin_amdgcn_wmma_f32_16x16x32_bf16(false, __builtin_bit_cast(v16bf, au),
                                                    false, __builtin_bit_cast(v16bf, bu),
                                                    (short)0, acc, false, false);
    }
    __syncthreads();
  }
#pragma unroll
  for (int r = 0; r < 8; ++r) {
    int c = 16 * wave + 8 * half + r;
    mel[(size_t)(pos0 + n) * NMEL_ + c] = acc[r] + pbias[c];
  }
}

extern "C" void kernel_launch(void* const* d_in, const int* in_sizes, int n_in,
                              void* d_out, int out_size, void* d_ws, size_t ws_size,
                              hipStream_t stream) {
  (void)in_sizes; (void)n_in; (void)out_size; (void)ws_size;

  const float* enc_out   = (const float*)d_in[0];
  const float* enc_mask  = (const float*)d_in[1];
  const float* pitch_tgt = (const float*)d_in[2];
  const int*   durations = (const int*)d_in[3];
  const float* dur_c0_w = (const float*)d_in[4],  *dur_c0_b = (const float*)d_in[5];
  const float* dur_n0_g = (const float*)d_in[6],  *dur_n0_b = (const float*)d_in[7];
  const float* dur_c1_w = (const float*)d_in[8],  *dur_c1_b = (const float*)d_in[9];
  const float* dur_n1_g = (const float*)d_in[10], *dur_n1_b = (const float*)d_in[11];
  const float* dur_fc_w = (const float*)d_in[12], *dur_fc_b = (const float*)d_in[13];
  const float* pit_c0_w = (const float*)d_in[14], *pit_c0_b = (const float*)d_in[15];
  const float* pit_n0_g = (const float*)d_in[16], *pit_n0_b = (const float*)d_in[17];
  const float* pit_c1_w = (const float*)d_in[18], *pit_c1_b = (const float*)d_in[19];
  const float* pit_n1_g = (const float*)d_in[20], *pit_n1_b = (const float*)d_in[21];
  const float* pit_fc_w = (const float*)d_in[22], *pit_fc_b = (const float*)d_in[23];
  const float* pemb_w   = (const float*)d_in[24], *pemb_b   = (const float*)d_in[25];
  const float* proj_w   = (const float*)d_in[26], *proj_b   = (const float*)d_in[27];

  // output regions (flat concat, f32)
  float* out       = (float*)d_out;
  float* o_mel     = out;                              // 16*2048*80
  float* o_mask    = out + 2621440;                    // 16*2048
  float* o_durpred = out + 2654208;                    // 16*256
  float* o_logdur  = out + 2658304;                    // 16*256
  float* o_pitch   = out + 2662400;                    // 16*256

  // workspace carve-out (256B aligned)
  size_t off = 0;
  char* base = (char*)d_ws;
  auto alloc = [&](size_t bytes) -> void* {
    void* p = base + off;
    off += (bytes + 255) & ~(size_t)255;
    return p;
  };
  u16* X0     = (u16*)alloc((size_t)B_ * D_ * T_ * 2);          // enc transposed+masked
  u16* durW0  = (u16*)alloc((size_t)FILT_ * D_ * 3 * 2);
  u16* pitW0  = (u16*)alloc((size_t)FILT_ * D_ * 3 * 2);
  u16* durW1  = (u16*)alloc((size_t)FILT_ * FILT_ * 3 * 2);
  u16* pitW1  = (u16*)alloc((size_t)FILT_ * FILT_ * 3 * 2);
  u16* projWb = (u16*)alloc((size_t)NMEL_ * D_ * 2);
  u16* H0d    = (u16*)alloc((size_t)B_ * FILT_ * T_ * 2);
  u16* H0p    = (u16*)alloc((size_t)B_ * FILT_ * T_ * 2);
  u16* H1d    = (u16*)alloc((size_t)B_ * FILT_ * T_ * 2);
  u16* H1p    = (u16*)alloc((size_t)B_ * FILT_ * T_ * 2);
  u16* encb   = (u16*)alloc((size_t)B_ * T_ * D_ * 2);
  int* cums   = (int*)alloc((size_t)B_ * T_ * 4);
  int* declen = (int*)alloc((size_t)B_ * 4);
  int* idxm   = (int*)alloc((size_t)B_ * M_ * 4);

  // weight conversions to bf16 (conv weights permuted to block-major k ordering)
  k_cvt_w<<<(294912 + 255) / 256, 256, 0, stream>>>(dur_c0_w, durW0, D_, 294912);
  k_cvt_w<<<(294912 + 255) / 256, 256, 0, stream>>>(pit_c0_w, pitW0, D_, 294912);
  k_cvt_w<<<(196608 + 255) / 256, 256, 0, stream>>>(dur_c1_w, durW1, FILT_, 196608);
  k_cvt_w<<<(196608 + 255) / 256, 256, 0, stream>>>(pit_c1_w, pitW1, FILT_, 196608);
  k_cvt<<<(30720 + 255) / 256, 256, 0, stream>>>(proj_w, projWb, 30720);

  // predictor pipeline
  k_transpose_mask<<<(B_ * D_ * T_) / 256, 256, 0, stream>>>(enc_out, enc_mask, X0);
  k_conv_ln<<<256, 512, 0, stream>>>(X0, durW0, dur_c0_b, dur_n0_g, dur_n0_b, H0d, D_);
  k_conv_ln<<<256, 512, 0, stream>>>(X0, pitW0, pit_c0_b, pit_n0_g, pit_n0_b, H0p, D_);
  k_conv_ln<<<256, 512, 0, stream>>>(H0d, durW1, dur_c1_b, dur_n1_g, dur_n1_b, H1d, FILT_);
  k_conv_ln<<<256, 512, 0, stream>>>(H0p, pitW1, pit_c1_b, pit_n1_g, pit_n1_b, H1p, FILT_);
  k_fc<<<512, 256, 0, stream>>>(H1d, dur_fc_w, dur_fc_b, enc_mask, o_logdur, o_durpred, 1);
  k_fc<<<512, 256, 0, stream>>>(H1p, pit_fc_w, pit_fc_b, enc_mask, o_pitch, nullptr, 0);

  // length regulation + mel projection
  k_encb<<<(B_ * T_ * D_) / 256, 256, 0, stream>>>(enc_out, pitch_tgt, pemb_w, pemb_b, encb);
  k_scan<<<B_, 256, 0, stream>>>(durations, cums, declen);
  k_idx<<<(B_ * M_) / 256, 256, 0, stream>>>(cums, declen, idxm, o_mask);
  k_mel<<<(B_ * M_) / 16, 160, 0, stream>>>(encb, projWb, proj_b, idxm, o_mel);
}
